// MultiBoxLoss_45243185496281
// MI455X (gfx1250) — compile-verified, matrix operand order checked by reading, not run on hardware
//
#include <hip/hip_runtime.h>
#include <hip/hip_bf16.h>

#define BB 64
#define PP 8732
#define TT 16
#define CC 81
#define TILES 546            // ceil(8732/16)

typedef __attribute__((ext_vector_type(16))) _Float16 v16h;
typedef __attribute__((ext_vector_type(8)))  float    v8f;

// ---------------- zero init ----------------
__global__ void mbox_zero(int* num_pos, float* accum, int* total_pos) {
    int t = threadIdx.x;
    if (t < BB) num_pos[t] = 0;
    if (t < 4)  accum[t] = 0.f;
    if (t == 0) *total_pos = 0;
}

// ---------------- matching + localization loss ----------------
__global__ __launch_bounds__(1024) void mbox_match(
    const float* __restrict__ loc_data,  // [B,P,4]
    const float* __restrict__ priors,    // [P,4] center form
    const float* __restrict__ truths,    // [B,T,4] corner form
    const int*   __restrict__ labels,    // [B,T]
    int* __restrict__ conf_t, int* __restrict__ num_pos,
    float* __restrict__ accum, int* __restrict__ total_pos)
{
    const int b   = blockIdx.x;
    const int tid = threadIdx.x;
    __shared__ float s_tr[TT][4];
    __shared__ float s_area[TT];
    __shared__ int   s_lab[TT];
    __shared__ unsigned long long s_best[TT];
    __shared__ int   s_bp[TT];

    if (tid < TT) {
        const float* t4 = truths + (b * TT + tid) * 4;
        float x0 = t4[0], y0 = t4[1], x1 = t4[2], y1 = t4[3];
        s_tr[tid][0] = x0; s_tr[tid][1] = y0; s_tr[tid][2] = x1; s_tr[tid][3] = y1;
        s_area[tid]  = (x1 - x0) * (y1 - y0);
        s_lab[tid]   = labels[b * TT + tid];
        s_best[tid]  = 0ull;
    }
    __syncthreads();

    constexpr int NPR = (PP + 1023) / 1024;   // 9 priors per thread
    float ovr[NPR]; int idxr[NPR];
    unsigned long long lb[TT];
#pragma unroll
    for (int t = 0; t < TT; ++t) lb[t] = 0ull;

    for (int j = 0; j < NPR; ++j) {
        int p = tid + j * 1024;
        ovr[j] = -1.f; idxr[j] = 0;
        if (p < PP) {
            float cx = priors[p*4+0], cy = priors[p*4+1];
            float w  = priors[p*4+2], h  = priors[p*4+3];
            float px0 = cx - 0.5f*w, py0 = cy - 0.5f*h;
            float px1 = cx + 0.5f*w, py1 = cy + 0.5f*h;
            float ap = w * h;
            float bo = -1.f; int bi = 0;
#pragma unroll
            for (int t = 0; t < TT; ++t) {
                float ix0 = fmaxf(s_tr[t][0], px0), iy0 = fmaxf(s_tr[t][1], py0);
                float ix1 = fminf(s_tr[t][2], px1), iy1 = fminf(s_tr[t][3], py1);
                float iw = fmaxf(ix1 - ix0, 0.f), ih = fmaxf(iy1 - iy0, 0.f);
                float inter = iw * ih;
                float ov = inter / (s_area[t] + ap - inter);
                if (ov > bo) { bo = ov; bi = t; }   // first-max ties -> smallest t
                // pack: ov bits high, inverted prior idx low -> argmax prefers smallest p
                unsigned long long pk =
                    (((unsigned long long)__float_as_uint(ov)) << 32) |
                    (unsigned)(0xFFFFFFFFu - (unsigned)p);
                if (pk > lb[t]) lb[t] = pk;
            }
            ovr[j] = bo; idxr[j] = bi;
        }
    }
#pragma unroll
    for (int t = 0; t < TT; ++t) atomicMax(&s_best[t], lb[t]);
    __syncthreads();
    if (tid < TT)
        s_bp[tid] = (int)(0xFFFFFFFFu - (unsigned)(s_best[tid] & 0xFFFFFFFFull));
    __syncthreads();

    float l1 = 0.f; int npos = 0;
    for (int j = 0; j < NPR; ++j) {
        int p = tid + j * 1024;
        if (p >= PP) continue;
        float ov = ovr[j]; int ti = idxr[j];
        // override: every gt claims its best prior; duplicate scatter -> last t wins
#pragma unroll
        for (int t = TT - 1; t >= 0; --t)
            if (s_bp[t] == p) { ov = 2.f; ti = t; break; }
        int c = (ov < 0.5f) ? 0 : (s_lab[ti] + 1);
        conf_t[b * PP + p] = c;
        if (c > 0) {
            ++npos;
            float cx = priors[p*4+0], cy = priors[p*4+1];
            float w  = priors[p*4+2], h  = priors[p*4+3];
            float mx0 = s_tr[ti][0], my0 = s_tr[ti][1];
            float mx1 = s_tr[ti][2], my1 = s_tr[ti][3];
            float gx = ((mx0 + mx1) * 0.5f - cx) / (0.1f * w);
            float gy = ((my0 + my1) * 0.5f - cy) / (0.1f * h);
            float gw = __logf((mx1 - mx0) / w) / 0.2f;
            float gh = __logf((my1 - my0) / h) / 0.2f;
            const float* ld = loc_data + ((long)b * PP + p) * 4;
            float d, a;
            d = ld[0] - gx; a = fabsf(d); l1 += (a < 1.f) ? 0.5f*d*d : a - 0.5f;
            d = ld[1] - gy; a = fabsf(d); l1 += (a < 1.f) ? 0.5f*d*d : a - 0.5f;
            d = ld[2] - gw; a = fabsf(d); l1 += (a < 1.f) ? 0.5f*d*d : a - 0.5f;
            d = ld[3] - gh; a = fabsf(d); l1 += (a < 1.f) ? 0.5f*d*d : a - 0.5f;
        }
    }
    // wave32 reduce, then one atomic per wave
#pragma unroll
    for (int o = 16; o > 0; o >>= 1) {
        l1   += __shfl_down(l1, o, 32);
        npos += __shfl_down(npos, o, 32);
    }
    if ((tid & 31) == 0) {
        if (l1 != 0.f) atomicAdd(&accum[0], l1);
        if (npos) { atomicAdd(&num_pos[b], npos); atomicAdd(total_pos, npos); }
    }
}

// ---------------- cross-entropy via WMMA row-sum of exps ----------------
// One wave per 16-prior tile. A = exp(conf - rowmax) as 16x32 f16 (CDNA5 A layout),
// B = ones 32x16 -> D row-sums; 3 chained wmma cover classes 0..95 (81 valid).
// Loads are vectorized: each lane's classes form runs of 8 contiguous floats.
__global__ __launch_bounds__(256) void mbox_ce(
    const float* __restrict__ conf_data,  // [B,P,C]
    const int*   __restrict__ conf_t,     // [B,P]
    float* __restrict__ ce, float* __restrict__ accum)
{
    int wave = (blockIdx.x * blockDim.x + threadIdx.x) >> 5;
    int lane = threadIdx.x & 31;
    int b    = wave / TILES;
    int tile = wave % TILES;
    int m    = lane & 15;     // matrix row
    int half = lane >> 4;     // K-half per A layout
    int p    = tile * 16 + m;
    bool valid = p < PP;
    const float* row = conf_data + ((long)b * PP + p) * CC;

    // this lane's 48 classes: chunk c, elem e -> k = 32c + 8*half + (e<8 ? e : e+8)
    float vals[3][16];
    if (valid) {
        // chunks 0,1 and chunk 2 group 0: fully in-range runs of 8 -> 2x b128 each
#pragma unroll
        for (int c = 0; c < 3; ++c)
#pragma unroll
            for (int grp = 0; grp < 2; ++grp) {
                int g = 32 * c + half * 8 + 16 * grp;   // first class of the run
                if (c == 2 && grp == 1) continue;       // tail handled below
                float4 q0 = *(const float4*)(row + g);
                float4 q1 = *(const float4*)(row + g + 4);
                int e0 = grp * 8;
                vals[c][e0+0] = q0.x; vals[c][e0+1] = q0.y;
                vals[c][e0+2] = q0.z; vals[c][e0+3] = q0.w;
                vals[c][e0+4] = q1.x; vals[c][e0+5] = q1.y;
                vals[c][e0+6] = q1.z; vals[c][e0+7] = q1.w;
            }
        // chunk 2 group 1: classes 80+8h..87+8h, only k<81 valid
#pragma unroll
        for (int e = 8; e < 16; ++e) {
            int k = 64 + half * 8 + 8 + e;
            vals[2][e] = (k < CC) ? row[k] : -1e30f;
        }
    } else {
#pragma unroll
        for (int c = 0; c < 3; ++c)
#pragma unroll
            for (int e = 0; e < 16; ++e) vals[c][e] = -1e30f;
    }

    float lmax = -1e30f;
#pragma unroll
    for (int c = 0; c < 3; ++c)
#pragma unroll
        for (int e = 0; e < 16; ++e) lmax = fmaxf(lmax, vals[c][e]);
    float rmax = fmaxf(lmax, __shfl_xor(lmax, 16, 32));  // combine K-halves

    v16h ones;
#pragma unroll
    for (int i = 0; i < 16; ++i) ones[i] = (_Float16)1.0f;
    v8f acc = {};
#pragma unroll
    for (int c = 0; c < 3; ++c) {
        v16h a;
#pragma unroll
        for (int e = 0; e < 16; ++e) {
            // padded / invalid entries hold -1e30 -> exp underflows, force exact 0
            float x = vals[c][e] - rmax;
            float ev = (x > -80.f) ? __expf(x) : 0.f;
            a[e] = (_Float16)ev;
        }
        acc = __builtin_amdgcn_wmma_f32_16x16x32_f16(
            false, a, false, ones, (short)0, acc, false, false);
    }
    // D layout: lane L, VGPR r holds row (r + 8*(L>>4)); columns identical (B=ones)
    int s7 = lane & 7;
    float myslot = acc[0];
#pragma unroll
    for (int i = 1; i < 8; ++i) myslot = (s7 == i) ? acc[i] : myslot;
    // lane owning rowsum(m): lane (m&7) | ((m>>3)<<4)
    int src = (m & 7) | (((m >> 3) & 1) << 4);
    float sumexp = __shfl(myslot, src, 32);

    // both halves compute cev (same p); only half==0 writes/contributes
    float contrib = 0.f;
    if (valid) {
        int tgt = conf_t[b * PP + p];
        float cv = row[tgt];
        float cev = rmax + __logf(sumexp) - cv;   // logsumexp - conf[target]
        if (half == 0) {
            ce[b * PP + p] = cev;
            if (tgt > 0) contrib = cev;
        }
    }
    // one atomic per wave for the positive-CE sum
#pragma unroll
    for (int o = 16; o > 0; o >>= 1) contrib += __shfl_down(contrib, o, 32);
    if (lane == 0 && contrib != 0.f) atomicAdd(&accum[1], contrib);
}

// ---------------- hard-negative mining: exact top-K via bitwise bisection ----------------
__global__ __launch_bounds__(256) void mbox_select(
    const float* __restrict__ ce, const int* __restrict__ conf_t,
    const int* __restrict__ num_pos, float* __restrict__ accum)
{
    int b = blockIdx.x, tid = threadIdx.x;
    __shared__ int s_cnt;
    __shared__ float s_sum;
    int np = num_pos[b];
    int K = min(3 * np, PP - 1);
    if (K <= 0) return;                       // uniform across block
    const float* cep = ce + (long)b * PP;
    const int*   cfp = conf_t + (long)b * PP;

    // find v_K = K-th largest of loss_c (nonneg floats: bits are order-isomorphic)
    unsigned lo = 0u, hi = 0x7F7FFFFFu;
    while (lo < hi) {
        unsigned mid = lo + (hi - lo + 1u) / 2u;
        float mf = __uint_as_float(mid);
        int c = 0;
        for (int p = tid; p < PP; p += 256) {
            float v = (cfp[p] > 0) ? 0.f : cep[p];
            if (v >= mf) ++c;
        }
        if (tid == 0) s_cnt = 0;
        __syncthreads();
        atomicAdd(&s_cnt, c);
        __syncthreads();
        int total = s_cnt;
        __syncthreads();
        if (total >= K) lo = mid; else hi = mid - 1u;
    }
    float vK = __uint_as_float(lo);
    int cgt = 0; float sgt = 0.f;
    for (int p = tid; p < PP; p += 256) {
        float v = (cfp[p] > 0) ? 0.f : cep[p];
        if (v > vK) { ++cgt; sgt += v; }
    }
    if (tid == 0) { s_cnt = 0; s_sum = 0.f; }
    __syncthreads();
    atomicAdd(&s_cnt, cgt);
    atomicAdd(&s_sum, sgt);
    __syncthreads();
    if (tid == 0)
        atomicAdd(&accum[2], s_sum + (float)(K - s_cnt) * vK);
}

// ---------------- finalize ----------------
__global__ void mbox_finalize(const float* __restrict__ accum,
                              const int* __restrict__ total_pos,
                              float* __restrict__ out)
{
    float N = fmaxf((float)(*total_pos), 1.f);
    out[0] = accum[0] / N;
    out[1] = (accum[1] + accum[2]) / N;
}

extern "C" void kernel_launch(void* const* d_in, const int* in_sizes, int n_in,
                              void* d_out, int out_size, void* d_ws, size_t ws_size,
                              hipStream_t stream) {
    const float* loc_data  = (const float*)d_in[0];   // [64,8732,4]
    const float* conf_data = (const float*)d_in[1];   // [64,8732,81]
    const float* priors    = (const float*)d_in[2];   // [8732,4]
    const float* truths    = (const float*)d_in[3];   // [64,16,4]
    const int*   labels    = (const int*)  d_in[4];   // [64,16]
    float* out = (float*)d_out;

    const size_t BP4 = (size_t)BB * PP * 4;           // 2,235,392 bytes
    char* ws = (char*)d_ws;
    int*   conf_t    = (int*)  (ws);
    float* ce        = (float*)(ws + BP4);
    int*   num_pos   = (int*)  (ws + 2 * BP4);
    float* accum     = (float*)(ws + 2 * BP4 + 256);
    int*   total_pos = (int*)  (ws + 2 * BP4 + 256 + 16);

    mbox_zero<<<1, 64, 0, stream>>>(num_pos, accum, total_pos);
    mbox_match<<<BB, 1024, 0, stream>>>(loc_data, priors, truths, labels,
                                        conf_t, num_pos, accum, total_pos);
    // 64*546 waves, 8 waves (256 thr) per block
    mbox_ce<<<(BB * TILES) / 8, 256, 0, stream>>>(conf_data, conf_t, ce, accum);
    mbox_select<<<BB, 256, 0, stream>>>(ce, conf_t, num_pos, accum);
    mbox_finalize<<<1, 1, 0, stream>>>(accum, total_pos, out);
}
